// BasicRnn_31645319037179
// MI455X (gfx1250) — matrix operand order repaired and checked by reading, not compiled, not run on hardware
//
#include <hip/hip_runtime.h>

// Problem dimensions (match reference).
#define B_SZ  64
#define SEQ   512
#define DIN   1024
#define DH    2048
#define DOUT  1024
#define FUT   16
#define KH    (DIN + DH)   // 3072

typedef __bf16 bf16_t;
typedef __attribute__((ext_vector_type(16))) __bf16 v16bf;
typedef __attribute__((ext_vector_type(8)))  __bf16 v8bf;
typedef __attribute__((ext_vector_type(8)))  float  v8f;
typedef __attribute__((ext_vector_type(4)))  float  v4f;

union Frag16 { v16bf v; v8bf h[2]; };

// Async global->LDS copy: 16 bytes per lane, no destination VGPRs, tracked
// by ASYNCcnt (CDNA5 ISA). ldsoff = workgroup LDS byte addr.
#define ASYNC_CP16(ldsoff, gptr)                                              \
    asm volatile("global_load_async_to_lds_b128 %0, %1, off"                  \
                 :: "v"(ldsoff), "v"(gptr) : "memory")

// Partial waits on ASYNCcnt: pipeline depth enforced by the counter, not by
// the register allocator. Async loads complete in order, so these are exact.
#define ASYNC_WAIT_20() asm volatile("s_wait_asynccnt 0x14" ::: "memory")
#define ASYNC_WAIT_10() asm volatile("s_wait_asynccnt 0xa"  ::: "memory")
#define ASYNC_WAIT_0()  asm volatile("s_wait_asynccnt 0x0"  ::: "memory")

// ---------------------------------------------------------------------------
// Weight packing: fp32 row-major [K, N] -> bf16 WMMA B-fragments.
// Fragment (n_tile, k_tile) holds B[k_tile*32 .. +31][n_tile*16 .. +15].
// Lane l holds column n = l%16 and K values (v/8)*16 + (l/16)*8 + v%8, v = 0..15,
// K-major packed in 16 bf16 (32 contiguous bytes per lane).
// ---------------------------------------------------------------------------
__global__ __launch_bounds__(256) void pack_weights_bf16(
    const float* __restrict__ W, bf16_t* __restrict__ P, int K, int N)
{
    long idx = (long)blockIdx.x * 256 + threadIdx.x;
    long total = (long)K * N;
    if (idx >= total) return;
    const int KT = K / 32;
    long frag = idx >> 9;
    int within = (int)(idx & 511);
    int lane = within >> 4;
    int v    = within & 15;
    int nt   = (int)(frag / KT);
    int kt   = (int)(frag % KT);
    int k = kt * 32 + (v >> 3) * 16 + (lane >> 4) * 8 + (v & 7);
    int n = nt * 16 + (lane & 15);
    P[idx] = (bf16_t)W[(long)k * N + n];
}

// Bulk fp32 -> bf16 conversion (8 elems / thread); once per launch, parallel,
// OFF the sequential dependency chain.
__global__ __launch_bounds__(256) void convert_f32_bf16(
    const float* __restrict__ src, bf16_t* __restrict__ dst, long n8)
{
    long i = (long)blockIdx.x * 256 + threadIdx.x;
    if (i >= n8) return;
    const v4f* q = (const v4f*)(src + i * 8);
    v4f f0 = q[0], f1 = q[1];
    v8bf r;
    #pragma unroll
    for (int j = 0; j < 4; ++j) { r[j] = (bf16_t)f0[j]; r[4 + j] = (bf16_t)f1[j]; }
    *(v8bf*)(dst + i * 8) = r;
}

__global__ __launch_bounds__(256) void zero_bf16(bf16_t* __restrict__ p, int n)
{
    int i = blockIdx.x * 256 + threadIdx.x;
    if (i < n) p[i] = (bf16_t)0.0f;
}

__device__ __forceinline__ void mma4(v8f* acc, const Frag16& a, const Frag16* b)
{
    #pragma unroll
    for (int nt = 0; nt < 4; ++nt)
        acc[nt] = __builtin_amdgcn_wmma_f32_16x16x32_bf16(
            false, a.v, false, b[nt].v, (short)0, acc[nt], false, false);
}

// ---------------------------------------------------------------------------
// Async-pipelined GEMM region. Per-wave private 3-stage LDS ring:
//   stage = [ A-frag 1KB | B-frag x4 4KB ] = 5KB; ring = 15KB/wave.
// Producer issues 10 async b128 copies per K-chunk, 3 chunks ahead; consumer
// does s_wait_asynccnt (2 stages in flight -> partial wait), ds_load_b128
// fragment reads, then 4 WMMAs. Prefetch depth costs LDS, not VGPRs, so the
// scheduler cannot collapse the pipeline. Ring slot kt%3 is reissued for
// kt+3 only after chunk kt's ds_loads were consumed (program order + long
// global latency on the refill makes the overwrite race-free).
// NCH >= 4 required (32 / 64 here).
template <int NCH>
__device__ __forceinline__ void gemm_lds(v8f* acc, const bf16_t* abase,
                                         const bf16_t* bp, long nstr,
                                         char* swave, int lane)
{
    const unsigned sw_off = (unsigned)(size_t)swave;  // LDS byte address
    const int loff32 = lane * 32;
    const int loff16 = lane * 16;

    auto issue = [&](int kt) {
        unsigned sb = sw_off + (unsigned)((kt % 3) * 5120);
        const bf16_t* ag = abase + (long)kt * 32;
        ASYNC_CP16(sb + loff32,      ag);       // A run0 (16B)
        ASYNC_CP16(sb + loff32 + 16, ag + 16);  // A run1 (16B)
        #pragma unroll
        for (int nt = 0; nt < 4; ++nt) {
            const bf16_t* bg = bp + (long)kt * 512 + (long)nt * nstr + loff16;
            unsigned tb = sb + 1024 + nt * 1024;
            ASYNC_CP16(tb + loff32,      bg);
            ASYNC_CP16(tb + loff32 + 16, bg + 8);
        }
    };
    auto consume = [&](int kt) {
        const char* st = swave + (kt % 3) * 5120;
        Frag16 a;
        a.h[0] = *(const v8bf*)(st + loff32);
        a.h[1] = *(const v8bf*)(st + loff32 + 16);
        Frag16 b[4];
        #pragma unroll
        for (int nt = 0; nt < 4; ++nt) {
            const char* tb = st + 1024 + nt * 1024;
            b[nt].h[0] = *(const v8bf*)(tb + loff32);
            b[nt].h[1] = *(const v8bf*)(tb + loff32 + 16);
        }
        mma4(acc, a, b);
    };

    issue(0); issue(1); issue(2);
    #pragma nounroll
    for (int kt = 0; kt < NCH - 3; ++kt) {
        ASYNC_WAIT_20();          // stage kt resident; kt+1, kt+2 in flight
        consume(kt);
        issue(kt + 3);
    }
    ASYNC_WAIT_20(); consume(NCH - 3);
    ASYNC_WAIT_10(); consume(NCH - 2);
    ASYNC_WAIT_0();  consume(NCH - 1);
}

// ---------------------------------------------------------------------------
// Hidden step: h_out = tanh( concat(x_t, h_in) @ Wh + bh ).
// x region (32 chunks) + h region (64 chunks). Grid: 16 blocks x 256 thr;
// wave w: m_tile = w&3, 4 n_tiles at blockIdx*8 + (w>>2)*4.
// ---------------------------------------------------------------------------
__global__ __launch_bounds__(256) void rnn_hidden_step(
    const bf16_t* __restrict__ x_src, long x_stride,   // bf16, row stride elems
    const bf16_t* __restrict__ h_in,                   // [64, DH] bf16
    const bf16_t* __restrict__ WhP,
    const float*  __restrict__ bh,
    bf16_t*       __restrict__ h_out)
{
    __shared__ __align__(16) char smem[8 * 3 * 5120];  // 120 KB (<= 320 KB WGP LDS)

    const int lane    = threadIdx.x & 31;
    const int wave    = threadIdx.x >> 5;
    const int m_tile  = wave & 3;
    const int n_tile0 = blockIdx.x * 8 + (wave >> 2) * 4;
    const int half    = lane >> 4;
    const int mrow    = m_tile * 16 + (lane & 15);
    constexpr int KT   = KH / 32;          // 96
    constexpr int KTX  = DIN / 32;         // 32
    constexpr int KTH  = DH / 32;          // 64
    constexpr long NSTR = (long)KT * 512;  // packed n_tile stride (elems)

    char* swave = smem + wave * (3 * 5120);
    v8f acc[4] = {};

    const bf16_t* bpx = WhP + (long)n_tile0 * NSTR + lane * 16 - lane * 16;  // frag base
    // (lane offset applied inside gemm_lds; keep bp at fragment origin)
    bpx = WhP + (long)n_tile0 * NSTR;
    const bf16_t* bph = bpx + (long)KTX * 512;

    gemm_lds<KTX>(acc, x_src + (long)mrow * x_stride + half * 8, bpx, NSTR, swave, lane);
    gemm_lds<KTH>(acc, h_in + (long)mrow * DH + half * 8,        bph, NSTR, swave, lane);

    // Epilogue: bias + tanh -> bf16. D layout: VGPR r -> M = half*8 + r,
    // N = lane & 15.
    #pragma unroll
    for (int nt = 0; nt < 4; ++nt) {
        const int n = (n_tile0 + nt) * 16 + (lane & 15);
        const float bias = bh[n];
        #pragma unroll
        for (int r = 0; r < 8; ++r) {
            const int m = m_tile * 16 + half * 8 + r;
            h_out[(long)m * DH + n] = (bf16_t)tanhf(acc[nt][r] + bias);
        }
    }
}

// ---------------------------------------------------------------------------
// Output step: out = h @ Wo + bo. WRITE_FUT mirrors the result as bf16 into
// fut_out for the autoregressive feedback path. Grid: 8 blocks x 256 thr.
// ---------------------------------------------------------------------------
template <bool WRITE_FUT>
__global__ __launch_bounds__(256) void rnn_out_step(
    const bf16_t* __restrict__ h,
    const bf16_t* __restrict__ WoP,
    const float*  __restrict__ bo,
    float*        __restrict__ out, long out_stride,
    bf16_t*       __restrict__ fut_out)
{
    __shared__ __align__(16) char smem[8 * 3 * 5120];

    const int lane    = threadIdx.x & 31;
    const int wave    = threadIdx.x >> 5;
    const int m_tile  = wave & 3;
    const int n_tile0 = blockIdx.x * 8 + (wave >> 2) * 4;
    const int half    = lane >> 4;
    const int mrow    = m_tile * 16 + (lane & 15);
    constexpr int KT   = DH / 32;          // 64
    constexpr long NSTR = (long)KT * 512;

    char* swave = smem + wave * (3 * 5120);
    v8f acc[4] = {};

    const bf16_t* bp = WoP + (long)n_tile0 * NSTR;
    gemm_lds<KT>(acc, h + (long)mrow * DH + half * 8, bp, NSTR, swave, lane);

    #pragma unroll
    for (int nt = 0; nt < 4; ++nt) {
        const int n = (n_tile0 + nt) * 16 + (lane & 15);
        const float bias = bo[n];
        #pragma unroll
        for (int r = 0; r < 8; ++r) {
            const int m = m_tile * 16 + half * 8 + r;
            const float v = acc[nt][r] + bias;
            out[(long)m * out_stride + n] = v;
            if (WRITE_FUT) fut_out[(long)m * DOUT + n] = (bf16_t)v;
        }
    }
}

// ---------------------------------------------------------------------------
// Host side: pack weights + convert x (parallel, off the chain), zero h0,
// then stream-ordered sequential scan: 512 (hidden, out) pairs + 16
// autoregressive pairs. Graph capture replays the whole chain.
// ---------------------------------------------------------------------------
extern "C" void kernel_launch(void* const* d_in, const int* in_sizes, int n_in,
                              void* d_out, int out_size, void* d_ws, size_t ws_size,
                              hipStream_t stream)
{
    (void)in_sizes; (void)n_in; (void)out_size; (void)ws_size;
    const float* x  = (const float*)d_in[0];   // [B, S, DIN] fp32
    const float* Wh = (const float*)d_in[1];   // [KH, DH]
    const float* bh = (const float*)d_in[2];   // [DH]
    const float* Wo = (const float*)d_in[3];   // [DH, DOUT]
    const float* bo = (const float*)d_in[4];   // [DOUT]
    float* out = (float*)d_out;

    char* ws = (char*)d_ws;
    // Workspace (bytes): WhP 12582912 | WoP 4194304 | h_a 262144 | h_b 262144 |
    // futx 131072 | x_bf 67108864  => ~84.5 MB total.
    bf16_t* WhP  = (bf16_t*)(ws + 0);
    bf16_t* WoP  = (bf16_t*)(ws + 12582912);
    bf16_t* h_a  = (bf16_t*)(ws + 16777216);
    bf16_t* h_b  = (bf16_t*)(ws + 17039360);
    bf16_t* futx = (bf16_t*)(ws + 17301504);
    bf16_t* x_bf = (bf16_t*)(ws + 17432576);

    {
        long tot = (long)KH * DH;
        pack_weights_bf16<<<(int)((tot + 255) / 256), 256, 0, stream>>>(Wh, WhP, KH, DH);
        tot = (long)DH * DOUT;
        pack_weights_bf16<<<(int)((tot + 255) / 256), 256, 0, stream>>>(Wo, WoP, DH, DOUT);
        long n8 = (long)B_SZ * SEQ * DIN / 8;
        convert_f32_bf16<<<(int)((n8 + 255) / 256), 256, 0, stream>>>(x, x_bf, n8);
        zero_bf16<<<(B_SZ * DH + 255) / 256, 256, 0, stream>>>(h_a, B_SZ * DH);
    }

    bf16_t* hc = h_a;
    bf16_t* hn = h_b;
    for (int t = 0; t < SEQ; ++t) {
        rnn_hidden_step<<<DH / 128, 256, 0, stream>>>(
            x_bf + (long)t * DIN, (long)SEQ * DIN, hc, WhP, bh, hn);
        rnn_out_step<false><<<DOUT / 128, 256, 0, stream>>>(
            hn, WoP, bo, out + (long)t * DOUT, (long)SEQ * DOUT, nullptr);
        bf16_t* tmp = hc; hc = hn; hn = tmp;
    }

    float* pred = out + (long)B_SZ * SEQ * DOUT;
    for (int ft = 0; ft < FUT; ++ft) {
        const bf16_t* xin = (ft == 0) ? (x_bf + (long)(SEQ - 1) * DIN) : futx;
        const long    xst = (ft == 0) ? (long)SEQ * DIN : (long)DIN;
        rnn_hidden_step<<<DH / 128, 256, 0, stream>>>(xin, xst, hc, WhP, bh, hn);
        rnn_out_step<true><<<DOUT / 128, 256, 0, stream>>>(
            hn, WoP, bo, pred + (long)ft * DOUT, (long)FUT * DOUT, futx);
        bf16_t* tmp = hc; hc = hn; hn = tmp;
    }
}
